// _MambaBlock_55061480734890
// MI455X (gfx1250) — matrix operand order repaired and checked by reading, not compiled, not run on hardware
//
#include <hip/hip_runtime.h>
#include <hip/hip_bf16.h>
#include <math.h>

// ---------------- model constants ----------------
constexpr int DM    = 1024;           // d_model
constexpr int DI    = 2048;           // d_inner
constexpr int H     = 32;             // heads
constexpr int P     = 64;             // head dim
constexpr int G     = 128;            // mimo_rank * d_state
constexpr int DXBC  = DI + 2*H*G;     // 10240
constexpr int DPROJ = 2*DI + 2*H*G + H; // 12320
constexpr int CH    = 128;            // chunk
constexpr int NC    = 16;             // chunks per sequence
constexpr int Bsz   = 2;
constexpr int Lseq  = 2048;
constexpr int BL    = Bsz * Lseq;     // 4096 tokens
constexpr int DCONV = 4;

// ---------------- WMMA plumbing ----------------
typedef __attribute__((ext_vector_type(16))) __bf16 v16bf;
typedef __attribute__((ext_vector_type(8)))  __bf16 v8bf;
typedef __attribute__((ext_vector_type(8)))  float  v8f;

union Frag { v16bf v; v8bf h[2]; };

// A fragment 16x32 bf16: lane holds one row (lane&15); elems 0..7 -> k = kbase + (lane>>4)*8 + j,
// elems 8..15 -> k = kbase + 16 + (lane>>4)*8 + j  (CDNA5 ISA 7.12.2)
__device__ inline v16bf load_a_frag(const __bf16* row, int kbase, int half) {
  Frag f;
  f.h[0] = *(const v8bf*)(row + kbase + half * 8);
  f.h[1] = *(const v8bf*)(row + kbase + 16 + half * 8);
  return f.v;
}
// B fragment 32x16 bf16 (stored as Bw[N][K], row = output col): lane col = lane&15,
// elems 0..7 -> k = kbase + (lane>>4)*8 + j, elems 8..15 -> k = kbase + 16 + (lane>>4)*8 + j
__device__ inline v16bf load_b_frag(const __bf16* row, int kbase, int half) {
  Frag f;
  const __bf16* p = row + kbase + half * 8;
  f.h[0] = *(const v8bf*)(p);
  f.h[1] = *(const v8bf*)(p + 16);
  return f.v;
}
__device__ inline v8f wmma_bf16(v16bf a, v16bf b, v8f c) {
  return __builtin_amdgcn_wmma_f32_16x16x32_bf16(false, a, false, b, (short)0, c, false, false);
}

// ---------------- CDNA5 async copy global -> LDS (ASYNCcnt) ----------------
// 16 bytes per lane. LDS operand = addr[31:0] of the flat shared pointer.
__device__ inline void async_copy16(const void* gsrc, void* ldst) {
  unsigned lds = (unsigned)(unsigned long long)ldst;
  unsigned long long ga = (unsigned long long)gsrc;
  asm volatile("global_load_async_to_lds_b128 %0, %1, off"
               :: "v"(lds), "v"(ga) : "memory");
}
__device__ inline void wait_async0() {
  asm volatile("s_wait_asynccnt 0x0" ::: "memory");
}

// ---------------- elementwise / norm kernels ----------------
__global__ void cast_f32_bf16(const float* __restrict__ src, __bf16* __restrict__ dst, long n) {
  long i = (long)blockIdx.x * blockDim.x + threadIdx.x;
  long stride = (long)gridDim.x * blockDim.x;
  for (; i < n; i += stride) dst[i] = (__bf16)src[i];
}

__global__ void rmsnorm_cast(const float* __restrict__ x, const float* __restrict__ w,
                             __bf16* __restrict__ out, int cols) {
  int row = blockIdx.x;
  const float* xr = x + (long)row * cols;
  float ss = 0.f;
  for (int i = threadIdx.x; i < cols; i += blockDim.x) { float v = xr[i]; ss += v * v; }
  __shared__ float red[256];
  red[threadIdx.x] = ss; __syncthreads();
  for (int off = 128; off > 0; off >>= 1) {
    if (threadIdx.x < off) red[threadIdx.x] += red[threadIdx.x + off];
    __syncthreads();
  }
  float rstd = rsqrtf(red[0] / cols + 1e-5f);
  for (int i = threadIdx.x; i < cols; i += blockDim.x)
    out[(long)row * cols + i] = (__bf16)(xr[i] * w[i] * rstd);
}

// ------- generic WMMA GEMM, copy-free double-buffered pipeline --------------
// C[M,N] = A[M,K] * Bw[N,K]^T (+resid).  Requires K % 64 == 0 (both call sites).
__global__ __launch_bounds__(256) void gemm_bf16(
    const __bf16* __restrict__ A, const __bf16* __restrict__ Bw,
    __bf16* __restrict__ Cb, float* __restrict__ Cf, const float* __restrict__ resid,
    int M, int N, int K, int lda, int ldb, int ldc) {
  const int lane = threadIdx.x & 31;
  const int wave = threadIdx.x >> 5;
  const int half = lane >> 4;
  const int l16  = lane & 15;
  const int wm = wave & 3;                 // 4 waves along M
  const int wn = wave >> 2;                // 2 waves along N
  const int mbase = blockIdx.y * 64 + wm * 16;
  const int nbase = blockIdx.x * 128 + wn * 64;

  int arow = mbase + l16; if (arow > M - 1) arow = M - 1;
  const __bf16* Arow = A + (long)arow * lda;
  const __bf16* Brow[4];
#pragma unroll
  for (int t = 0; t < 4; ++t) {
    int n = nbase + t * 16 + l16; if (n > N - 1) n = N - 1;
    Brow[t] = Bw + (long)n * ldb;
  }

  v8f acc[4] = {};
  // buffer 0 holds fragments for step k, buffer 1 for step k+32; roles alternate
  v16bf a0 = load_a_frag(Arow, 0, half);
  v16bf b0[4];
#pragma unroll
  for (int t = 0; t < 4; ++t) b0[t] = load_b_frag(Brow[t], 0, half);

  int k = 0;
  for (; k + 64 < K; k += 64) {
    v16bf a1 = load_a_frag(Arow, k + 32, half);
    v16bf b1[4];
#pragma unroll
    for (int t = 0; t < 4; ++t) b1[t] = load_b_frag(Brow[t], k + 32, half);
    __builtin_prefetch(Arow + k + 192, 0, 0);       // global_prefetch_b8
#pragma unroll
    for (int t = 0; t < 4; ++t) acc[t] = wmma_bf16(a0, b0[t], acc[t]);

    a0 = load_a_frag(Arow, k + 64, half);
#pragma unroll
    for (int t = 0; t < 4; ++t) b0[t] = load_b_frag(Brow[t], k + 64, half);
#pragma unroll
    for (int t = 0; t < 4; ++t) acc[t] = wmma_bf16(a1, b1[t], acc[t]);
  }
  {  // tail: k = K - 64, two remaining 32-wide steps
    v16bf a1 = load_a_frag(Arow, k + 32, half);
    v16bf b1[4];
#pragma unroll
    for (int t = 0; t < 4; ++t) b1[t] = load_b_frag(Brow[t], k + 32, half);
#pragma unroll
    for (int t = 0; t < 4; ++t) acc[t] = wmma_bf16(a0, b0[t], acc[t]);
#pragma unroll
    for (int t = 0; t < 4; ++t) acc[t] = wmma_bf16(a1, b1[t], acc[t]);
  }

  // C layout: VGPR r, lane -> M = mbase + (lane>>4)*8 + r, N = nbase + t*16 + (lane&15)
  for (int t = 0; t < 4; ++t) {
    int n = nbase + t * 16 + l16;
    if (n >= N) continue;
    for (int r = 0; r < 8; ++r) {
      int m = mbase + half * 8 + r;
      if (m >= M) continue;
      long idx = (long)m * ldc + n;
      float v = acc[t][r];
      if (Cf) Cf[idx] = v + (resid ? resid[idx] : 0.f);
      else    Cb[idx] = (__bf16)v;
    }
  }
}

// ---------------- conv / dt / cumsum ----------------
__global__ void conv_silu(const __bf16* __restrict__ zx, const float* __restrict__ cw,
                          const float* __restrict__ cb, __bf16* __restrict__ xbc) {
  long id = (long)blockIdx.x * blockDim.x + threadIdx.x;
  long n = (long)BL * DXBC;
  if (id >= n) return;
  int c  = (int)(id % DXBC);
  int bt = (int)(id / DXBC);
  int t  = bt % Lseq;
  float acc = cb[c];
  for (int k = 0; k < DCONV; ++k) {
    int tt = t - (DCONV - 1) + k;
    if (tt >= 0)
      acc += (float)zx[(long)(bt - (DCONV - 1) + k) * DPROJ + DI + c] * cw[c * DCONV + k];
  }
  float s = acc / (1.f + __expf(-acc));
  xbc[id] = (__bf16)s;
}

__global__ void dt_softplus(const __bf16* __restrict__ zx, const float* __restrict__ bias,
                            float* __restrict__ dt) {
  int id = blockIdx.x * blockDim.x + threadIdx.x;
  if (id >= BL * H) return;
  int h = id % H, bt = id / H;
  float x = (float)zx[(long)bt * DPROJ + DI + DXBC + h] + bias[h];
  dt[id] = (x > 20.f) ? x : log1pf(__expf(x));
}

__global__ void chunk_cumsum(const float* __restrict__ dt, const float* __restrict__ A_log,
                             float* __restrict__ cum) {
  int h = blockIdx.x % H;
  int c = (blockIdx.x / H) % NC;
  int b = blockIdx.x / (H * NC);
  int l = threadIdx.x;
  float Ah = -__expf(A_log[h]);
  int bt = b * Lseq + c * CH + l;
  __shared__ float sc[CH];
  sc[l] = dt[(long)bt * H + h] * Ah;
  __syncthreads();
  for (int off = 1; off < CH; off <<= 1) {
    float t = (l >= off) ? sc[l - off] : 0.f;
    __syncthreads();
    sc[l] += t;
    __syncthreads();
  }
  cum[(long)bt * H + h] = sc[l];
}

// ---------------- intra-chunk SSM: one workgroup per (b, chunk, head) ----------------
__global__ __launch_bounds__(256) void ssm_intra(
    const __bf16* __restrict__ xbc, const float* __restrict__ dt,
    const float* __restrict__ cum, const float* __restrict__ Dsk,
    float* __restrict__ S, float* __restrict__ ybuf) {
  const int h = blockIdx.x % H;
  const int c = (blockIdx.x / H) % NC;
  const int b = blockIdx.x / (H * NC);
  const int tid = threadIdx.x;
  const int lane = tid & 31, wave = tid >> 5, half = lane >> 4, l16 = lane & 15;

  __shared__ __align__(16) __bf16 sC[CH][G + 8];     // Cc row-major (l,g)
  __shared__ __align__(16) __bf16 sBS[CH][G + 8];    // Bc, then re-used for bf16 scores
  __shared__ __align__(16) __bf16 sBwT[G][CH + 8];   // (w*Bc)^T (g,l)
  __shared__ __align__(16) __bf16 sXT[P][CH + 8];    // Xc^T (p,l)
  __shared__ float scum[CH], sdt[CH], sw[CH];

  const long rowbase = (long)b * Lseq + (long)c * CH;

  // async row-major copies: Cc and Bc tiles (16B per lane-op, ASYNCcnt tracked)
  for (int idx = tid; idx < CH * (G / 8); idx += 256) {
    int l = idx >> 4, g8 = (idx & 15) * 8;
    long row = (rowbase + l) * DXBC;
    async_copy16(xbc + row + DI + H * G + h * G + g8, &sC[l][g8]);
    async_copy16(xbc + row + DI + h * G + g8,         &sBS[l][g8]);
  }

  if (tid < CH) {
    float cv = cum[(rowbase + tid) * H + h];
    float dv = dt[(rowbase + tid) * H + h];
    float clast = cum[(rowbase + CH - 1) * H + h];
    scum[tid] = cv; sdt[tid] = dv;
    sw[tid] = __expf(clast - cv) * dv;
  }
  __syncthreads();   // sw visible for the transpose staging below

  // transposed stages overlap with in-flight async copies
  for (int idx = tid; idx < CH * G; idx += 256) {
    int g = idx % G, l = idx / G;
    float bv = (float)xbc[(rowbase + l) * DXBC + DI + h * G + g];
    sBwT[g][l] = (__bf16)(bv * sw[l]);
  }
  for (int idx = tid; idx < CH * P; idx += 256) {
    int p = idx % P, l = idx / P;
    sXT[p][l] = xbc[(rowbase + l) * DXBC + h * P + p];
  }
  wait_async0();
  __syncthreads();

  // scores[l,s] = sum_g C[l,g]*B[s,g]; wave handles l-tile = wave, all 8 s-tiles
  v8f acc[8] = {};
  {
    const __bf16* Arow = &sC[wave * 16 + l16][0];
    for (int k = 0; k < G; k += 32) {
      v16bf a = load_a_frag(Arow, k, half);
#pragma unroll
      for (int t = 0; t < 8; ++t) {
        v16bf bfr = load_b_frag(&sBS[t * 16 + l16][0], k, half);
        acc[t] = wmma_bf16(a, bfr, acc[t]);
      }
    }
  }
  __syncthreads();   // all reads of Bc done; safe to overwrite sBS with scores
  for (int t = 0; t < 8; ++t) {
    int s = t * 16 + l16;
    for (int r = 0; r < 8; ++r) {
      int l = wave * 16 + half * 8 + r;
      float v = 0.f;
      if (s <= l) v = acc[t][r] * __expf(scum[l] - scum[s]) * sdt[s];
      sBS[l][s] = (__bf16)v;
    }
  }
  __syncthreads();

  // Y_intra = scores @ X  (+ D_skip * X)
  {
    v8f yacc[4] = {};
    const __bf16* Arow = &sBS[wave * 16 + l16][0];
    for (int k = 0; k < CH; k += 32) {
      v16bf a = load_a_frag(Arow, k, half);
#pragma unroll
      for (int t = 0; t < 4; ++t) {
        v16bf bfr = load_b_frag(&sXT[t * 16 + l16][0], k, half);
        yacc[t] = wmma_bf16(a, bfr, yacc[t]);
      }
    }
    float Dh = Dsk[h];
    for (int t = 0; t < 4; ++t) {
      int p = t * 16 + l16;
      for (int r = 0; r < 8; ++r) {
        int l = wave * 16 + half * 8 + r;
        float xv = (float)sXT[p][l];
        ybuf[(rowbase + l) * DI + h * P + p] = yacc[t][r] + Dh * xv;
      }
    }
  }

  // S_chunk[g,p] = sum_l w[l]*B[l,g]*X[l,p]
  {
    v8f sacc[4] = {};
    const __bf16* Arow = &sBwT[wave * 16 + l16][0];
    for (int k = 0; k < CH; k += 32) {
      v16bf a = load_a_frag(Arow, k, half);
#pragma unroll
      for (int t = 0; t < 4; ++t) {
        v16bf bfr = load_b_frag(&sXT[t * 16 + l16][0], k, half);
        sacc[t] = wmma_bf16(a, bfr, sacc[t]);
      }
    }
    long Sbase = ((long)(b * NC + c) * H + h) * (long)(G * P);
    for (int t = 0; t < 4; ++t) {
      int p = t * 16 + l16;
      for (int r = 0; r < 8; ++r) {
        int g = wave * 16 + half * 8 + r;
        S[Sbase + (long)g * P + p] = sacc[t][r];
      }
    }
  }
}

// ---------------- inter-chunk state recurrence (16 sequential steps) ----------------
__global__ void chunk_scan(const float* __restrict__ S, const float* __restrict__ cum,
                           __bf16* __restrict__ HpT) {
  long id = (long)blockIdx.x * blockDim.x + threadIdx.x;
  if (id >= (long)Bsz * H * P * G) return;
  int g = (int)(id % G);
  int p = (int)((id / G) % P);
  int h = (int)((id / ((long)G * P)) % H);
  int b = (int)(id / ((long)G * P * H));
  float state = 0.f;
  for (int c = 0; c < NC; ++c) {
    long base = (long)(b * NC + c) * H + h;
    HpT[base * (long)(P * G) + (long)p * G + g] = (__bf16)state;   // state BEFORE chunk c
    float dl = __expf(cum[((long)b * Lseq + (long)c * CH + (CH - 1)) * H + h]);
    state = dl * state + S[base * (long)(G * P) + (long)g * P + p];
  }
}

// ---------------- Y_inter = (C * exp(cum)) @ Hprev ----------------
__global__ __launch_bounds__(256) void ssm_inter(
    const __bf16* __restrict__ xbc, const float* __restrict__ cum,
    const __bf16* __restrict__ HpT, float* __restrict__ ybuf) {
  const int h = blockIdx.x % H;
  const int c = (blockIdx.x / H) % NC;
  const int b = blockIdx.x / (H * NC);
  const int tid = threadIdx.x;
  const int lane = tid & 31, wave = tid >> 5, half = lane >> 4, l16 = lane & 15;
  __shared__ __align__(16) __bf16 sC[CH][G + 8];
  __shared__ __align__(16) __bf16 sHT[P][G + 8];
  __shared__ float scum[CH];

  const long rowbase = (long)b * Lseq + (long)c * CH;
  long Hbase = ((long)(b * NC + c) * H + h) * (long)(P * G);

  // async row-major copies of both operands
  for (int idx = tid; idx < CH * (G / 8); idx += 256) {
    int l = idx >> 4, g8 = (idx & 15) * 8;
    async_copy16(xbc + (rowbase + l) * DXBC + DI + H * G + h * G + g8, &sC[l][g8]);
  }
  for (int idx = tid; idx < P * (G / 8); idx += 256) {
    int p = idx >> 4, g8 = (idx & 15) * 8;
    async_copy16(HpT + Hbase + (long)p * G + g8, &sHT[p][g8]);
  }
  if (tid < CH) scum[tid] = cum[(rowbase + tid) * H + h];
  wait_async0();
  __syncthreads();

  v8f acc[4] = {};
  const __bf16* Arow = &sC[wave * 16 + l16][0];
  for (int k = 0; k < G; k += 32) {
    v16bf a = load_a_frag(Arow, k, half);
#pragma unroll
    for (int t = 0; t < 4; ++t) {
      v16bf bfr = load_b_frag(&sHT[t * 16 + l16][0], k, half);
      acc[t] = wmma_bf16(a, bfr, acc[t]);
    }
  }
  // exp(cum[l]) depends only on the row -> pulled out of the GEMM
  for (int t = 0; t < 4; ++t) {
    int p = t * 16 + l16;
    for (int r = 0; r < 8; ++r) {
      int l = wave * 16 + half * 8 + r;
      long idx = (rowbase + l) * DI + h * P + p;
      ybuf[idx] += __expf(scum[l]) * acc[t][r];
    }
  }
}

// ---------------- gated RMSNorm ----------------
__global__ void gated_rmsnorm(const float* __restrict__ ybuf, const __bf16* __restrict__ zx,
                              const float* __restrict__ gw, __bf16* __restrict__ out) {
  int bt = blockIdx.x;
  float ss = 0.f;
  for (int i = threadIdx.x; i < DI; i += blockDim.x) {
    float z = (float)zx[(long)bt * DPROJ + i];
    float v = ybuf[(long)bt * DI + i] * (z / (1.f + __expf(-z)));
    ss += v * v;
  }
  __shared__ float red[256];
  red[threadIdx.x] = ss; __syncthreads();
  for (int off = 128; off > 0; off >>= 1) {
    if (threadIdx.x < off) red[threadIdx.x] += red[threadIdx.x + off];
    __syncthreads();
  }
  float rstd = rsqrtf(red[0] / DI + 1e-5f);
  for (int i = threadIdx.x; i < DI; i += blockDim.x) {
    float z = (float)zx[(long)bt * DPROJ + i];
    float v = ybuf[(long)bt * DI + i] * (z / (1.f + __expf(-z)));
    out[(long)bt * DI + i] = (__bf16)(v * gw[i] * rstd);
  }
}

// ---------------- host orchestration ----------------
extern "C" void kernel_launch(void* const* d_in, const int* in_sizes, int n_in,
                              void* d_out, int out_size, void* d_ws, size_t ws_size,
                              hipStream_t stream) {
  const float* x       = (const float*)d_in[0];
  const float* norm_w  = (const float*)d_in[1];
  const float* in_w    = (const float*)d_in[2];
  const float* conv_w  = (const float*)d_in[3];
  const float* conv_b  = (const float*)d_in[4];
  const float* dt_bias = (const float*)d_in[5];
  const float* A_log   = (const float*)d_in[6];
  const float* D_skip  = (const float*)d_in[7];
  const float* gnorm_w = (const float*)d_in[8];
  const float* out_w   = (const float*)d_in[9];
  float* out = (float*)d_out;
  (void)in_sizes; (void)n_in; (void)out_size; (void)ws_size;

  char* ws = (char*)d_ws;
  size_t off = 0;
  auto alloc = [&](size_t bytes) { char* p = ws + off; off = (off + bytes + 255) & ~(size_t)255; return p; };
  __bf16* xn   = (__bf16*)alloc((size_t)BL * DM * 2);
  __bf16* win  = (__bf16*)alloc((size_t)DPROJ * DM * 2);
  __bf16* wout = (__bf16*)alloc((size_t)DM * DI * 2);
  __bf16* zx   = (__bf16*)alloc((size_t)BL * DPROJ * 2);
  __bf16* xbc  = (__bf16*)alloc((size_t)BL * DXBC * 2);
  float*  dtb  = (float*)alloc((size_t)BL * H * 4);
  float*  cumb = (float*)alloc((size_t)BL * H * 4);
  float*  Sb   = (float*)alloc((size_t)Bsz * NC * H * G * P * 4);
  __bf16* HpT  = (__bf16*)alloc((size_t)Bsz * NC * H * P * G * 2);
  float*  ybuf = (float*)alloc((size_t)BL * DI * 4);
  __bf16* ybf  = (__bf16*)alloc((size_t)BL * DI * 2);

  cast_f32_bf16<<<4096, 256, 0, stream>>>(in_w, win, (long)DPROJ * DM);
  cast_f32_bf16<<<2048, 256, 0, stream>>>(out_w, wout, (long)DM * DI);
  rmsnorm_cast<<<BL, 256, 0, stream>>>(x, norm_w, xn, DM);

  {  // in_proj: zx[4096,12320] = xn[4096,1024] @ in_w[12320,1024]^T
    dim3 grid((DPROJ + 127) / 128, BL / 64);
    gemm_bf16<<<grid, 256, 0, stream>>>(xn, win, zx, nullptr, nullptr,
                                        BL, DPROJ, DM, DM, DM, DPROJ);
  }
  {
    long n = (long)BL * DXBC;
    conv_silu<<<(unsigned)((n + 255) / 256), 256, 0, stream>>>(zx, conv_w, conv_b, xbc);
  }
  dt_softplus<<<(BL * H + 255) / 256, 256, 0, stream>>>(zx, dt_bias, dtb);
  chunk_cumsum<<<Bsz * NC * H, CH, 0, stream>>>(dtb, A_log, cumb);
  ssm_intra<<<Bsz * NC * H, 256, 0, stream>>>(xbc, dtb, cumb, D_skip, Sb, ybuf);
  {
    long n = (long)Bsz * H * P * G;
    chunk_scan<<<(unsigned)((n + 255) / 256), 256, 0, stream>>>(Sb, cumb, HpT);
  }
  ssm_inter<<<Bsz * NC * H, 256, 0, stream>>>(xbc, cumb, HpT, ybuf);
  gated_rmsnorm<<<BL, 256, 0, stream>>>(ybuf, zx, gnorm_w, ybf);
  {  // out_proj + residual: out[4096,1024] = ybf @ out_w^T + x
    dim3 grid((DM + 127) / 128, BL / 64);
    gemm_bf16<<<grid, 256, 0, stream>>>(ybf, wout, nullptr, out, x,
                                        BL, DM, DI, DI, DI, DM);
  }
}
